// ApplyKmeans_63118839382467
// MI455X (gfx1250) — compile-verified, matrix operand order checked by reading, not run on hardware
//
#include <hip/hip_runtime.h>
#include <stdint.h>

#define DIMD 768
#define KCLUS 1024
#define ASTRIDE 776   // sA row stride in bf16: 1552B rows -> 16B-aligned, bank-conflict-free
#define BSTRIDE 136   // sBT row stride in bf16: 64 data DWORDs + 4 pad DWORDs (TDM pad)

typedef __attribute__((ext_vector_type(16))) __bf16 v16bf;
typedef __attribute__((ext_vector_type(8)))  float  v8f;
typedef __attribute__((ext_vector_type(4)))  unsigned int v4u;
typedef __attribute__((ext_vector_type(8)))  int v8i;
typedef __attribute__((ext_vector_type(4)))  int v4i;

#if defined(__AMDGCN__) && __has_builtin(__builtin_amdgcn_tensor_load_to_lds) && \
    __has_builtin(__builtin_amdgcn_s_wait_tensorcnt)
#define HAVE_TDM 1
#else
#define HAVE_TDM 0
#endif

union FragBF { v16bf v; uint4 q[2]; };

__device__ __forceinline__ unsigned short f2bf(float f) {
    unsigned int u = __float_as_uint(f);
    u += 0x7FFFu + ((u >> 16) & 1u);   // round-to-nearest-even
    return (unsigned short)(u >> 16);
}

// ---- prep: Ct[k][d] = C[d][k] (fp32) and Cbf[k][d] = bf16(C[d][k]) -------
__global__ __launch_bounds__(256) void transpose_C(const float* __restrict__ C,
                                                   float* __restrict__ Ct,
                                                   unsigned short* __restrict__ Cbf,
                                                   int writeCbf) {
    __shared__ float tile[32][33];
    const int bk = blockIdx.x * 32;           // centroid index
    const int bd = blockIdx.y * 32;           // feature index
    const int tx = threadIdx.x, ty = threadIdx.y;
    for (int j = 0; j < 32; j += 8)
        tile[ty + j][tx] = C[(long)(bd + ty + j) * KCLUS + bk + tx];
    __syncthreads();
    for (int j = 0; j < 32; j += 8) {
        const float v = tile[tx][ty + j];
        Ct[(long)(bk + ty + j) * DIMD + bd + tx] = v;
        if (writeCbf)
            Cbf[(long)(bk + ty + j) * DIMD + bd + tx] = f2bf(v);
    }
}

// ---- fused: bf16 WMMA GEMM + argmin + codeword gather --------------------
// Block: 256 threads (8 waves), 64 rows of x vs all 1024 centroids.
// Column tiles of 128; each wave owns a 32x32 macro-tile (2x2 WMMA tiles).
__global__ __launch_bounds__(256) void kmeans_assign_gather(
    const float* __restrict__ x, const float* __restrict__ Cf,
    const float* __restrict__ Cnorm, const float* __restrict__ Ct,
    const unsigned short* __restrict__ Cbf,
    int useCt, int useTdm, float* __restrict__ out)
{
    __shared__ unsigned short sA[64 * ASTRIDE];    // x tile bf16, padded (conflict-free)
    __shared__ unsigned short sBT[128 * BSTRIDE];  // C chunk, [n][k] padded
    __shared__ float sval[64 * 4];
    __shared__ int   sidx[64 * 4];
    __shared__ int   sids[64];

    const int tid  = threadIdx.x;
    const int lane = tid & 31;
    const int w    = tid >> 5;
    const int h    = lane >> 4;        // half of wave (K-split in 16-bit A/B layout)
    const int ln   = lane & 15;
    const int ms   = w >> 2;           // m-super: rows 32*ms..32*ms+31
    const int ns   = w & 3;            // n-super: 2 n-blocks within 128-col tile
    const int mb0  = 2 * ms, mb1 = 2 * ms + 1;
    const int nb0  = 2 * ns, nb1 = 2 * ns + 1;
    const long row0 = (long)blockIdx.x * 64;
#if HAVE_TDM
    const bool tdm = useTdm != 0;
#else
    const bool tdm = false;
#endif

    // per-lane fragment base offsets (elements); each fragment = two 16B runs
    const int aoff0 = (mb0 * 16 + ln) * ASTRIDE + 8 * h;
    const int aoff1 = (mb1 * 16 + ln) * ASTRIDE + 8 * h;
    const int boff0 = (nb0 * 16 + ln) * BSTRIDE + 8 * h;
    const int boff1 = (nb1 * 16 + ln) * BSTRIDE + 8 * h;

    // stage x tile (64 x 768 fp32 -> bf16 LDS), float4 loads / uint2 stores
    for (int i4 = tid; i4 < (64 * DIMD) / 4; i4 += 256) {
        const int e = i4 * 4;
        const int r = e / DIMD;
        const int c = e % DIMD;
        const float4 f = *(const float4*)(x + (row0 + r) * DIMD + c);
        uint2 p;
        p.x = (unsigned int)f2bf(f.x) | ((unsigned int)f2bf(f.y) << 16);
        p.y = (unsigned int)f2bf(f.z) | ((unsigned int)f2bf(f.w) << 16);
        *(uint2*)&sA[r * ASTRIDE + c] = p;
    }

    float bestv[16];
    int   besti[16];
#pragma unroll
    for (int i = 0; i < 16; ++i) { bestv[i] = 3.4e38f; besti[i] = 0; }

    for (int ntile = 0; ntile < KCLUS / 128; ++ntile) {
        const int n0 = ntile * 128;
        v8f acc[2][2];
        v8f zf = {};
        acc[0][0] = zf; acc[0][1] = zf; acc[1][0] = zf; acc[1][1] = zf;

        for (int dchunk = 0; dchunk < DIMD; dchunk += 128) {
            __syncthreads();
            if (tdm) {
#if HAVE_TDM
                // One wave drives the Tensor Data Mover: 2D tile 128 rows x 128
                // bf16 from Cbf[n0..n0+128)[dchunk..+128) into padded sBT.
                if (w == 0) {
                    const unsigned long long ga =
                        (unsigned long long)(const void*)(Cbf + (long)n0 * DIMD + dchunk);
                    const unsigned int ldsoff =
                        (unsigned int)(unsigned long long)(const void*)&sBT[0];
                    v4u g0;
                    g0.x = 1u;                               // count=1, user descriptor
                    g0.y = ldsoff;                           // lds_addr
                    g0.z = (unsigned int)(ga & 0xFFFFFFFFu); // global_addr[31:0]
                    g0.w = (unsigned int)((ga >> 32) & 0x01FFFFFFu) | (2u << 30); // type=2
                    v8i g1;
                    g1.s0 = (1 << 16)        // data_size = 2 bytes
                          | (1 << 20)        // pad_enable
                          | (5 << 22)        // pad_interval: 64 DWORDs (= 128 bf16 row)
                          | (3 << 25);       // pad_amount: 4 DWORDs -> stride 136 bf16
                    g1.s1 = (DIMD & 0xFFFF) << 16;                    // tensor_dim0 lo
                    g1.s2 = (DIMD >> 16) | ((KCLUS & 0xFFFF) << 16);  // dim0 hi | dim1 lo
                    g1.s3 = (KCLUS >> 16) | (128 << 16);              // dim1 hi | tile_dim0
                    g1.s4 = 128;                                      // tile_dim1 (tile_dim2=0)
                    g1.s5 = DIMD;                                     // tensor_dim0_stride lo
                    g1.s6 = 0;
                    g1.s7 = 0;
                    v4i gz = {0, 0, 0, 0};
                    v8i gz8 = {0, 0, 0, 0, 0, 0, 0, 0};
                    __builtin_amdgcn_tensor_load_to_lds(g0, g1, gz, gz, gz8, 0);
                    __builtin_amdgcn_s_wait_tensorcnt(0);
                }
                // prefetch next chunk of Cbf into L2 while this one computes
                if (dchunk + 128 < DIMD) {
                    const unsigned short* nxt = Cbf + (long)n0 * DIMD + dchunk + 128;
                    const int p = tid;       // 128 rows x 2 cachelines
                    __builtin_prefetch(nxt + (long)(p >> 1) * DIMD + (p & 1) * 64, 0, 0);
                }
#endif
            } else {
                // manual stage: C[dchunk..+128)[n0..+128) transposed into sBT[n][k]
                for (int i = tid; i < 128 * 32; i += 256) {
                    const int kk = i >> 5;
                    const int nn = (i & 31) * 4;
                    const float4 f = *(const float4*)(Cf + (long)(dchunk + kk) * KCLUS + n0 + nn);
                    sBT[(nn + 0) * BSTRIDE + kk] = f2bf(f.x);
                    sBT[(nn + 1) * BSTRIDE + kk] = f2bf(f.y);
                    sBT[(nn + 2) * BSTRIDE + kk] = f2bf(f.z);
                    sBT[(nn + 3) * BSTRIDE + kk] = f2bf(f.w);
                }
                if (dchunk + 128 < DIMD) {
                    const float* nxt = Cf + (long)(dchunk + 128) * KCLUS + n0;
                    for (int p = tid; p < 512; p += 256)      // 128 rows x 4 lines
                        __builtin_prefetch(nxt + (long)(p >> 2) * KCLUS + (p & 3) * 32, 0, 0);
                }
            }
            __syncthreads();

#pragma unroll
            for (int ks = 0; ks < 128; ks += 32) {
                const int k0 = dchunk + ks;
                FragBF a0, a1, b0, b1;
                // each fragment = two 16B-aligned b128 LDS loads
                a0.q[0] = *(const uint4*)&sA[aoff0 + k0];
                a0.q[1] = *(const uint4*)&sA[aoff0 + k0 + 16];
                a1.q[0] = *(const uint4*)&sA[aoff1 + k0];
                a1.q[1] = *(const uint4*)&sA[aoff1 + k0 + 16];
                b0.q[0] = *(const uint4*)&sBT[boff0 + ks];
                b0.q[1] = *(const uint4*)&sBT[boff0 + ks + 16];
                b1.q[0] = *(const uint4*)&sBT[boff1 + ks];
                b1.q[1] = *(const uint4*)&sBT[boff1 + ks + 16];
                acc[0][0] = __builtin_amdgcn_wmma_f32_16x16x32_bf16(
                                false, a0.v, false, b0.v, (short)0, acc[0][0], false, false);
                acc[0][1] = __builtin_amdgcn_wmma_f32_16x16x32_bf16(
                                false, a0.v, false, b1.v, (short)0, acc[0][1], false, false);
                acc[1][0] = __builtin_amdgcn_wmma_f32_16x16x32_bf16(
                                false, a1.v, false, b0.v, (short)0, acc[1][0], false, false);
                acc[1][1] = __builtin_amdgcn_wmma_f32_16x16x32_bf16(
                                false, a1.v, false, b1.v, (short)0, acc[1][1], false, false);
            }
        }

        // dist = Cnorm - 2*dot ; fold the wave's two n-tiles, then lane-butterfly min
        const int col0 = n0 + nb0 * 16 + ln;
        const int col1 = n0 + nb1 * 16 + ln;
        const float cn0 = Cnorm[col0];
        const float cn1 = Cnorm[col1];
#pragma unroll
        for (int tm = 0; tm < 2; ++tm) {
#pragma unroll
            for (int r = 0; r < 8; ++r) {
                float v0 = cn0 - 2.0f * acc[tm][0][r];
                float v1 = cn1 - 2.0f * acc[tm][1][r];
                float v; int idx;
                if (v1 < v0) { v = v1; idx = col1; } else { v = v0; idx = col0; }
#pragma unroll
                for (int m = 1; m < 16; m <<= 1) {
                    const float ov = __shfl_xor(v, m, 32);
                    const int   oi = __shfl_xor(idx, m, 32);
                    if (ov < v || (ov == v && oi < idx)) { v = ov; idx = oi; }
                }
                const int s = tm * 8 + r;
                if (v < bestv[s] || (v == bestv[s] && idx < besti[s])) {
                    bestv[s] = v; besti[s] = idx;
                }
            }
        }
    }

    // cross-wave argmin reduce (4 n-super groups per row)
    __syncthreads();
    if (ln == 0) {
#pragma unroll
        for (int tm = 0; tm < 2; ++tm) {
#pragma unroll
            for (int r = 0; r < 8; ++r) {
                const int row = (2 * ms + tm) * 16 + 8 * h + r;
                sval[row * 4 + ns] = bestv[tm * 8 + r];
                sidx[row * 4 + ns] = besti[tm * 8 + r];
            }
        }
    }
    __syncthreads();
    if (tid < 64) {
        float v = sval[tid * 4];
        int idx  = sidx[tid * 4];
#pragma unroll
        for (int j = 1; j < 4; ++j) {
            const float ov = sval[tid * 4 + j];
            const int   oi = sidx[tid * 4 + j];
            if (ov < v || (ov == v && oi < idx)) { v = ov; idx = oi; }
        }
        sids[tid] = idx;
    }
    __syncthreads();

    // gather codewords: out[row] = C.T[id] (float4-coalesced via Ct when available)
    if (useCt) {
        for (int i = tid; i < 64 * (DIMD / 4); i += 256) {
            const int r  = i / (DIMD / 4);
            const int d4 = i % (DIMD / 4);
            const int id = sids[r];
            const float4 f = *(const float4*)(Ct + (long)id * DIMD + d4 * 4);
            *(float4*)(out + (row0 + r) * DIMD + d4 * 4) = f;
        }
    } else {
        for (int r = 0; r < 64; ++r) {
            const int id = sids[r];
            float* dst = out + (row0 + r) * DIMD;
            for (int d = tid; d < DIMD; d += 256) dst[d] = Cf[(long)d * KCLUS + id];
        }
    }
}

extern "C" void kernel_launch(void* const* d_in, const int* in_sizes, int n_in,
                              void* d_out, int out_size, void* d_ws, size_t ws_size,
                              hipStream_t stream) {
    const float* x     = (const float*)d_in[0];   // [N, 768]
    const float* C     = (const float*)d_in[1];   // [768, 1024]
    const float* Cnorm = (const float*)d_in[2];   // [1, 1024]
    float* out = (float*)d_out;                   // [N, 768]

    const size_t ctBytes  = (size_t)KCLUS * DIMD * sizeof(float);           // 3 MB
    const size_t cbfBytes = (size_t)KCLUS * DIMD * sizeof(unsigned short);  // 1.5 MB
    float*          Ct  = (float*)d_ws;
    unsigned short* Cbf = (unsigned short*)((char*)d_ws + ctBytes);

    const int nrows  = in_sizes[0] / DIMD;        // 262144
    const int useCt  = (ws_size >= ctBytes) ? 1 : 0;
    const int useTdm = (ws_size >= ctBytes + cbfBytes) ? 1 : 0;

    if (useCt) {
        transpose_C<<<dim3(KCLUS / 32, DIMD / 32), dim3(32, 8), 0, stream>>>(C, Ct, Cbf, useTdm);
    }
    kmeans_assign_gather<<<nrows / 64, 256, 0, stream>>>(x, C, Cnorm, Ct, Cbf,
                                                         useCt, useTdm, out);
}
